// DeeperGCN_1838246002980
// MI455X (gfx1250) — matrix operand order repaired and checked by reading, not compiled, not run on hardware
//
#include <hip/hip_runtime.h>
#include <hip/hip_bf16.h>
#include <math.h>

// ---------------------------------------------------------------------------
// DeeperGCN (GENConv softmax aggregation, 6 layers) for MI455X / gfx1250.
// GEMMs run on v_wmma_f32_16x16x32_bf16 (activations+weights converted to
// bf16, f32 accumulate). Each wave computes a 16x64 output strip (4 N-tiles)
// so every A fragment load feeds 4 WMMAs; all 4 B fragments are loaded into
// distinct registers before the WMMA group so loads overlap matrix ops
// (partial s_wait_loadcnt instead of wait-0 before every WMMA).
// Edge softmax-aggregation is streaming float4 loads with f32 atomics
// (the HBM-bound part: ~8-10 GB total -> ~0.4 ms @ 23.3 TB/s).
// ---------------------------------------------------------------------------

#define NNODES   50000
#define NEDGES   600000
#define DIM      128
#define HID      256
#define NLAYERS  6
#define EPS_MSG  1e-7f
#define EPS_SM   1e-16f
#define LN_EPS   1e-5f

#define GF_RELU  1
#define GF_RESID 2

typedef __attribute__((ext_vector_type(16))) __bf16 v16bf;
typedef __attribute__((ext_vector_type(8)))  float  v8f;

static __device__ __forceinline__ unsigned short f32_to_bf16(float f) {
  unsigned int u = __float_as_uint(f);
  u += 0x7FFFu + ((u >> 16) & 1u);          // round-to-nearest-even
  return (unsigned short)(u >> 16);
}

static __device__ __forceinline__ void atomicMaxF32(float* addr, float val) {
  unsigned int* ia = (unsigned int*)addr;
  unsigned int old = __float_as_uint(*addr);
  while (__uint_as_float(old) < val) {
    unsigned int assumed = old;
    old = atomicCAS(ia, assumed, __float_as_uint(val));
    if (old == assumed) break;
  }
}

struct Frag32B { uint4 lo, hi; };

// ---------------------------------------------------------------------------
// bf16 WMMA GEMM: C[M,Nc] = A[M,K] @ B[K,Nc] (+bias) (+resid) (relu?)
// A: row-major bf16.  Bt: column-major bf16 (Bt[n*K + k]).
// One wave computes a 16x64 strip: 4 N-tiles sharing one A fragment per
// K-step (K-loop of 16x16x32 WMMAs). Requires Nc % 64 == 0, M % 16 == 0.
// Fragment layouts used here (contiguous k order, same for A and B):
//   A (16x32, 16-bit): lane m = lane&15; elems 0..15 -> K = hi16..hi16+15
//                      (hi16 = (lane>=16)*16)
//   B (32x16, 16-bit): lane n = lane&15; elems 0..15 -> K = hi16..hi16+15
//   C (16x16, f32):    lane n = lane&15; vgpr r -> M = r + (lane>=16)*8
// ---------------------------------------------------------------------------
__global__ __launch_bounds__(32)
void wmma_gemm_bf16(const unsigned short* __restrict__ A,
                    const unsigned short* __restrict__ Bt,
                    const float* __restrict__ bias,
                    const float* __restrict__ resid,
                    float* __restrict__ C,
                    int M, int K, int Nc, int flags)
{
  const int lane = threadIdx.x;
  const int bn   = blockIdx.x << 6;      // 64-wide N strip
  const int bm   = blockIdx.y << 4;
  const int m    = bm + (lane & 15);
  const int n0   = bn + (lane & 15);
  const int hi8  = (lane >> 4) << 3;     // 0 or 8
  const int hi16 = (lane >> 4) << 4;     // 0 or 16

  const unsigned short* arow = A + (size_t)m * K;
  const unsigned short* bcol[4];
#pragma unroll
  for (int j = 0; j < 4; ++j) bcol[j] = Bt + (size_t)(n0 + 16 * j) * K;

  v8f c[4];
#pragma unroll
  for (int j = 0; j < 4; ++j)
#pragma unroll
    for (int r = 0; r < 8; ++r) c[j][r] = 0.f;

  for (int kk = 0; kk < K; kk += 32) {
    // Issue ALL fragment loads first (distinct registers) so the scheduler
    // can overlap loads with the WMMA group via partial s_wait_loadcnt.
    Frag32B fa, fb0, fb1, fb2, fb3;
    fa.lo  = *reinterpret_cast<const uint4*>(arow + kk + hi16);
    fa.hi  = *reinterpret_cast<const uint4*>(arow + kk + hi16 + 8);
    fb0.lo = *reinterpret_cast<const uint4*>(bcol[0] + kk + hi16);
    fb0.hi = *reinterpret_cast<const uint4*>(bcol[0] + kk + hi16 + 8);
    fb1.lo = *reinterpret_cast<const uint4*>(bcol[1] + kk + hi16);
    fb1.hi = *reinterpret_cast<const uint4*>(bcol[1] + kk + hi16 + 8);
    fb2.lo = *reinterpret_cast<const uint4*>(bcol[2] + kk + hi16);
    fb2.hi = *reinterpret_cast<const uint4*>(bcol[2] + kk + hi16 + 8);
    fb3.lo = *reinterpret_cast<const uint4*>(bcol[3] + kk + hi16);
    fb3.hi = *reinterpret_cast<const uint4*>(bcol[3] + kk + hi16 + 8);

    v16bf a  = __builtin_bit_cast(v16bf, fa);
    v16bf b0 = __builtin_bit_cast(v16bf, fb0);
    v16bf b1 = __builtin_bit_cast(v16bf, fb1);
    v16bf b2 = __builtin_bit_cast(v16bf, fb2);
    v16bf b3 = __builtin_bit_cast(v16bf, fb3);

    c[0] = __builtin_amdgcn_wmma_f32_16x16x32_bf16(false, a, false, b0,
                                                   (short)0, c[0], false, false);
    c[1] = __builtin_amdgcn_wmma_f32_16x16x32_bf16(false, a, false, b1,
                                                   (short)0, c[1], false, false);
    c[2] = __builtin_amdgcn_wmma_f32_16x16x32_bf16(false, a, false, b2,
                                                   (short)0, c[2], false, false);
    c[3] = __builtin_amdgcn_wmma_f32_16x16x32_bf16(false, a, false, b3,
                                                   (short)0, c[3], false, false);
  }

  const int mtop = bm + hi8;
#pragma unroll
  for (int j = 0; j < 4; ++j) {
    const int nn = n0 + 16 * j;
    const float bv = bias ? bias[nn] : 0.f;
#pragma unroll
    for (int r = 0; r < 8; ++r) {
      int row = mtop + r;
      float v = c[j][r] + bv;
      if (flags & GF_RESID) v += resid[(size_t)row * Nc + nn];
      if (flags & GF_RELU)  v = fmaxf(v, 0.f);
      C[(size_t)row * Nc + nn] = v;
    }
  }
}

// ---------------------------------------------------------------------------
// Wave-per-row LayerNorm (+optional ReLU), emits f32 and/or bf16.
// cols is 128 or 256; each lane caches up to 8 elements in registers.
// ---------------------------------------------------------------------------
__global__ __launch_bounds__(256)
void ln_relu_kernel(const float* __restrict__ in, int rows, int cols,
                    const float* __restrict__ gamma,
                    const float* __restrict__ beta,
                    int do_relu,
                    float* __restrict__ out_f32,
                    unsigned short* __restrict__ out_bf16)
{
  int wave = (blockIdx.x * blockDim.x + threadIdx.x) >> 5;
  int lane = threadIdx.x & 31;
  if (wave >= rows) return;
  const float* row = in + (size_t)wave * cols;
  const int nchunk = cols >> 5;            // 4 or 8
  float v[8];
  float s = 0.f;
  for (int i = 0; i < nchunk; ++i) { v[i] = row[lane + (i << 5)]; s += v[i]; }
#pragma unroll
  for (int off = 16; off; off >>= 1) s += __shfl_xor(s, off, 32);
  float mu = s / (float)cols;
  float q = 0.f;
  for (int i = 0; i < nchunk; ++i) { float d = v[i] - mu; q += d * d; }
#pragma unroll
  for (int off = 16; off; off >>= 1) q += __shfl_xor(q, off, 32);
  float rinv = rsqrtf(q / (float)cols + LN_EPS);
  for (int i = 0; i < nchunk; ++i) {
    int col = lane + (i << 5);
    float val = (v[i] - mu) * rinv * gamma[col] + beta[col];
    if (do_relu) val = fmaxf(val, 0.f);
    if (out_f32)  out_f32[(size_t)wave * cols + col]  = val;
    if (out_bf16) out_bf16[(size_t)wave * cols + col] = f32_to_bf16(val);
  }
}

// ---------------------------------------------------------------------------
// Aggregation buffers init: mmax=-inf, denom=0, agg=0.
// ---------------------------------------------------------------------------
__global__ __launch_bounds__(256)
void agg_init_kernel(float* __restrict__ mmax, float* __restrict__ denom,
                     float* __restrict__ agg, long long n)
{
  long long i = (long long)blockIdx.x * blockDim.x + threadIdx.x;
  if (i >= n) return;
  mmax[i]  = -INFINITY;
  denom[i] = 0.f;
  agg[i]   = 0.f;
}

// ---------------------------------------------------------------------------
// Edge passes: 32 threads per edge, float4 per thread (D=128).
// msg = relu(g[src]) + eps ; scaled = msg * t
// ---------------------------------------------------------------------------
__global__ __launch_bounds__(256)
void edge_max_kernel(const float* __restrict__ g, const int* __restrict__ src,
                     const int* __restrict__ dst, const float* __restrict__ tarr,
                     int layer, float* __restrict__ mmax)
{
  long long tid = (long long)blockIdx.x * blockDim.x + threadIdx.x;
  if (tid >= (long long)NEDGES * 32) return;
  int e = (int)(tid >> 5);
  int c = ((int)tid & 31) << 2;
  float t = tarr[layer];
  int s = src[e], d = dst[e];
  const float4 gv = *reinterpret_cast<const float4*>(g + (size_t)s * DIM + c);
  float* mp = mmax + (size_t)d * DIM + c;
  atomicMaxF32(mp + 0, (fmaxf(gv.x, 0.f) + EPS_MSG) * t);
  atomicMaxF32(mp + 1, (fmaxf(gv.y, 0.f) + EPS_MSG) * t);
  atomicMaxF32(mp + 2, (fmaxf(gv.z, 0.f) + EPS_MSG) * t);
  atomicMaxF32(mp + 3, (fmaxf(gv.w, 0.f) + EPS_MSG) * t);
}

__global__ __launch_bounds__(256)
void edge_expsum_kernel(const float* __restrict__ g, const int* __restrict__ src,
                        const int* __restrict__ dst, const float* __restrict__ tarr,
                        int layer, const float* __restrict__ mmax,
                        float* __restrict__ denom)
{
  long long tid = (long long)blockIdx.x * blockDim.x + threadIdx.x;
  if (tid >= (long long)NEDGES * 32) return;
  int e = (int)(tid >> 5);
  int c = ((int)tid & 31) << 2;
  float t = tarr[layer];
  int s = src[e], d = dst[e];
  const float4 gv = *reinterpret_cast<const float4*>(g + (size_t)s * DIM + c);
  const float4 mv = *reinterpret_cast<const float4*>(mmax + (size_t)d * DIM + c);
  float* dp = denom + (size_t)d * DIM + c;
  atomicAdd(dp + 0, __expf((fmaxf(gv.x, 0.f) + EPS_MSG) * t - mv.x));
  atomicAdd(dp + 1, __expf((fmaxf(gv.y, 0.f) + EPS_MSG) * t - mv.y));
  atomicAdd(dp + 2, __expf((fmaxf(gv.z, 0.f) + EPS_MSG) * t - mv.z));
  atomicAdd(dp + 3, __expf((fmaxf(gv.w, 0.f) + EPS_MSG) * t - mv.w));
}

__global__ __launch_bounds__(256)
void edge_weighted_kernel(const float* __restrict__ g, const int* __restrict__ src,
                          const int* __restrict__ dst, const float* __restrict__ tarr,
                          int layer, const float* __restrict__ mmax,
                          const float* __restrict__ denom, float* __restrict__ agg)
{
  long long tid = (long long)blockIdx.x * blockDim.x + threadIdx.x;
  if (tid >= (long long)NEDGES * 32) return;
  int e = (int)(tid >> 5);
  int c = ((int)tid & 31) << 2;
  float t = tarr[layer];
  int s = src[e], d = dst[e];
  const float4 gv = *reinterpret_cast<const float4*>(g + (size_t)s * DIM + c);
  const float4 mv = *reinterpret_cast<const float4*>(mmax + (size_t)d * DIM + c);
  const float4 dv = *reinterpret_cast<const float4*>(denom + (size_t)d * DIM + c);
  float* ap = agg + (size_t)d * DIM + c;
  float m0 = fmaxf(gv.x, 0.f) + EPS_MSG, m1 = fmaxf(gv.y, 0.f) + EPS_MSG;
  float m2 = fmaxf(gv.z, 0.f) + EPS_MSG, m3 = fmaxf(gv.w, 0.f) + EPS_MSG;
  atomicAdd(ap + 0, m0 * (__expf(m0 * t - mv.x) / (dv.x + EPS_SM)));
  atomicAdd(ap + 1, m1 * (__expf(m1 * t - mv.y) / (dv.y + EPS_SM)));
  atomicAdd(ap + 2, m2 * (__expf(m2 * t - mv.z) / (dv.z + EPS_SM)));
  atomicAdd(ap + 3, m3 * (__expf(m3 * t - mv.w) / (dv.w + EPS_SM)));
}

// ---------------------------------------------------------------------------
// Small utility kernels.
// ---------------------------------------------------------------------------
__global__ __launch_bounds__(256)
void transpose_to_bf16(const float* __restrict__ W, unsigned short* __restrict__ Wt,
                       int K, int N)   // W[K,N] row-major -> Wt[N,K] (col-major)
{
  int idx = blockIdx.x * blockDim.x + threadIdx.x;
  if (idx >= K * N) return;
  int n = idx / K, k = idx - n * K;
  Wt[idx] = f32_to_bf16(W[(size_t)k * N + n]);
}

__global__ __launch_bounds__(256)
void convert_bf16(const float* __restrict__ in, unsigned short* __restrict__ out,
                  long long n)
{
  long long i = (long long)blockIdx.x * blockDim.x + threadIdx.x;
  if (i < n) out[i] = f32_to_bf16(in[i]);
}

__global__ __launch_bounds__(256)
void add_convert_kernel(const float* __restrict__ a, const float* __restrict__ b,
                        unsigned short* __restrict__ out, long long n)
{
  long long i = (long long)blockIdx.x * blockDim.x + threadIdx.x;
  if (i < n) out[i] = f32_to_bf16(a[i] + b[i]);
}

__global__ __launch_bounds__(256)
void head_kernel(const float* __restrict__ q, const float* __restrict__ w2,
                 const float* __restrict__ b2, float* __restrict__ out)
{
  int n = blockIdx.x * blockDim.x + threadIdx.x;
  if (n >= NNODES) return;
  const float* row = q + (size_t)n * DIM;
  float a0 = b2[0], a1 = b2[1];
#pragma unroll 4
  for (int k = 0; k < DIM; ++k) {
    float v = row[k];
    a0 += v * w2[2 * k];
    a1 += v * w2[2 * k + 1];
  }
  out[2 * n]     = a0;
  out[2 * n + 1] = a1;
}

// ---------------------------------------------------------------------------
extern "C" void kernel_launch(void* const* d_in, const int* in_sizes, int n_in,
                              void* d_out, int out_size, void* d_ws, size_t ws_size,
                              hipStream_t stream)
{
  const float* x      = (const float*)d_in[0];
  const int*   ei     = (const int*)d_in[1];     // [2,E] (src row, dst row)
  const float* enc_W  = (const float*)d_in[2];
  const float* enc_b  = (const float*)d_in[3];
  const float* ln_g   = (const float*)d_in[4];
  const float* ln_b   = (const float*)d_in[5];
  const float* tarr   = (const float*)d_in[6];
  const float* W1     = (const float*)d_in[7];
  const float* b1     = (const float*)d_in[8];
  const float* mln_g  = (const float*)d_in[9];
  const float* mln_b  = (const float*)d_in[10];
  const float* W2     = (const float*)d_in[11];
  const float* b2     = (const float*)d_in[12];
  const float* lin_W1 = (const float*)d_in[13];
  const float* lin_b1 = (const float*)d_in[14];
  const float* lin_W2 = (const float*)d_in[15];
  const float* lin_b2 = (const float*)d_in[16];
  float* out = (float*)d_out;

  const int* src = ei;
  const int* dst = ei + NEDGES;

  // ---- workspace layout (f32 unless noted) ----
  const size_t fN = (size_t)NNODES * DIM;            // 6.4M floats
  float* ws    = (float*)d_ws;
  float* h     = ws;                                 // [N,128] running state
  float* g     = ws + fN;                            // [N,128] LN+ReLU out / head tmp
  float* mmax  = ws + 2 * fN;                        // [N,128]
  float* denom = ws + 3 * fN;                        // [N,128]
  float* agg   = ws + 4 * fN;                        // [N,128]
  float* z     = mmax;                               // [N,256] aliases mmax+denom (dead)
  unsigned short* abf = (unsigned short*)(ws + 5 * fN);   // [N,256] bf16 staging
  unsigned short* encT = abf + (size_t)NNODES * HID;      // 128*128
  unsigned short* w1T  = encT + DIM * DIM;                // L * 256*128 (col-major)
  unsigned short* w2T  = w1T + (size_t)NLAYERS * DIM * HID; // L * 128*256
  unsigned short* linT = w2T + (size_t)NLAYERS * DIM * HID; // 128*128

  const int TP = 256;

  // ---- weight conversion (bf16, column-major for WMMA B fragments) ----
  transpose_to_bf16<<<(DIM * DIM + TP - 1) / TP, TP, 0, stream>>>(enc_W, encT, DIM, DIM);
  for (int i = 0; i < NLAYERS; ++i) {
    transpose_to_bf16<<<(DIM * HID + TP - 1) / TP, TP, 0, stream>>>(
        W1 + (size_t)i * DIM * HID, w1T + (size_t)i * DIM * HID, DIM, HID);
    transpose_to_bf16<<<(DIM * HID + TP - 1) / TP, TP, 0, stream>>>(
        W2 + (size_t)i * HID * DIM, w2T + (size_t)i * HID * DIM, HID, DIM);
  }
  transpose_to_bf16<<<(DIM * DIM + TP - 1) / TP, TP, 0, stream>>>(lin_W1, linT, DIM, DIM);

  // ---- encoder: h = x @ enc_W + enc_b ----
  convert_bf16<<<(int)((fN + TP - 1) / TP), TP, 0, stream>>>(x, abf, (long long)fN);
  {
    dim3 gd(DIM / 64, NNODES / 16);
    wmma_gemm_bf16<<<gd, 32, 0, stream>>>(abf, encT, enc_b, nullptr, h,
                                          NNODES, DIM, DIM, 0);
  }

  const int lnBlocks = (NNODES + 7) / 8;                 // 8 waves / 256-thread block
  const long long edgeWork = (long long)NEDGES * 32;
  const int eBlocks = (int)((edgeWork + TP - 1) / TP);
  const int nBlocks = (int)((fN + TP - 1) / TP);

  for (int i = 0; i < NLAYERS; ++i) {
    // g = relu(LN(h))
    ln_relu_kernel<<<lnBlocks, 256, 0, stream>>>(h, NNODES, DIM,
                                                 ln_g + i * DIM, ln_b + i * DIM,
                                                 1, g, nullptr);
    // segment softmax aggregation over edges
    agg_init_kernel<<<nBlocks, TP, 0, stream>>>(mmax, denom, agg, (long long)fN);
    edge_max_kernel<<<eBlocks, TP, 0, stream>>>(g, src, dst, tarr, i, mmax);
    edge_expsum_kernel<<<eBlocks, TP, 0, stream>>>(g, src, dst, tarr, i, mmax, denom);
    edge_weighted_kernel<<<eBlocks, TP, 0, stream>>>(g, src, dst, tarr, i, mmax, denom, agg);
    // out = agg + g  (bf16 for GEMM1 A operand)
    add_convert_kernel<<<nBlocks, TP, 0, stream>>>(agg, g, abf, (long long)fN);
    // z = out @ W1 + b1
    {
      dim3 gd(HID / 64, NNODES / 16);
      wmma_gemm_bf16<<<gd, 32, 0, stream>>>(abf, w1T + (size_t)i * DIM * HID,
                                            b1 + i * HID, nullptr, z,
                                            NNODES, DIM, HID, 0);
    }
    // z = relu(LN(z)) -> bf16
    ln_relu_kernel<<<lnBlocks, 256, 0, stream>>>(z, NNODES, HID,
                                                 mln_g + i * HID, mln_b + i * HID,
                                                 1, nullptr, abf);
    // h = h + (z @ W2 + b2)   (res+)
    {
      dim3 gd(DIM / 64, NNODES / 16);
      wmma_gemm_bf16<<<gd, 32, 0, stream>>>(abf, w2T + (size_t)i * HID * DIM,
                                            b2 + i * DIM, h, h,
                                            NNODES, HID, DIM, GF_RESID);
    }
  }

  // ---- head: q = relu(h @ lin_W1 + lin_b1); out = q @ lin_W2 + lin_b2 ----
  convert_bf16<<<nBlocks, TP, 0, stream>>>(h, abf, (long long)fN);
  {
    dim3 gd(DIM / 64, NNODES / 16);
    wmma_gemm_bf16<<<gd, 32, 0, stream>>>(abf, linT, lin_b1, nullptr, g,
                                          NNODES, DIM, DIM, GF_RELU);
  }
  head_kernel<<<(NNODES + TP - 1) / TP, TP, 0, stream>>>(g, lin_W2, lin_b2, out);

  (void)in_sizes; (void)n_in; (void)out_size; (void)ws_size;
}
